// Attention_20117626814893
// MI455X (gfx1250) — matrix operand order repaired and checked by reading, not compiled, not run on hardware
//
#include <hip/hip_runtime.h>
#include <hip/hip_bf16.h>

// Problem constants (from reference)
#define Bsz 32
#define Tt  2048
#define Hh  1024
#define LDd 1024
#define TCH 8          // T split for context partial sums

typedef __attribute__((ext_vector_type(16))) __bf16 v16bf;
typedef __attribute__((ext_vector_type(8)))  float  v8f;

// Branch-free fast tanh: tanh(x) = sign(x) * (1-e)/(1+e), e = exp(-2|x|).
// e in (0,1] -> no overflow, no special cases; v_exp_f32 is a TRANS op that
// co-executes with WMMA on CDNA5.
__device__ __forceinline__ float fast_tanh(float x) {
    float ax = __builtin_fabsf(x);
    float e  = __expf(-2.0f * ax);
    float r  = (1.0f - e) * __builtin_amdgcn_rcpf(1.0f + e);
    return __builtin_copysignf(r, x);
}

// ---------------------------------------------------------------------------
// Kernel 1: dec_w[b][l] = decoder_hidden[b] . W1[:,l] + b1[l]   (tiny GEMM)
// ---------------------------------------------------------------------------
__global__ void k_decw(const float* __restrict__ dec,
                       const float* __restrict__ W1,
                       const float* __restrict__ b1,
                       float* __restrict__ dec_w) {
    int idx = blockIdx.x * blockDim.x + threadIdx.x;   // B*LD threads
    int b = idx >> 10;
    int l = idx & (LDd - 1);
    float acc = b1[l];
    const float* dh = dec + b * Hh;
    for (int h = 0; h < Hh; ++h)
        acc += dh[h] * W1[h * LDd + l];                // coalesced over l
    dec_w[idx] = acc;
}

// ---------------------------------------------------------------------------
// Kernel 2: W2T[l][h] = bf16(W2[h][l])  — LDS-tiled transpose + convert
// ---------------------------------------------------------------------------
__global__ void k_w2t(const float* __restrict__ W2, __bf16* __restrict__ W2T) {
    __shared__ __bf16 tile[32][33];                    // +1 pad: no bank conflicts
    int l0 = (blockIdx.x & 31) * 32;
    int h0 = (blockIdx.x >> 5) * 32;
    int tx = threadIdx.x & 31, ty = threadIdx.x >> 5;  // 32 x 8
#pragma unroll
    for (int r = 0; r < 32; r += 8)
        tile[ty + r][tx] = (__bf16)W2[(h0 + ty + r) * LDd + l0 + tx];
    __syncthreads();
#pragma unroll
    for (int r = 0; r < 32; r += 8)
        W2T[(l0 + ty + r) * Hh + h0 + tx] = tile[tx][ty + r];
}

// ---------------------------------------------------------------------------
// Kernel 3: fused  enc_w = enc@W2+b2 ;  score = tanh(dec_w*enc_w)@V + bV
//   WG = 128 threads (4 waves), 64 rows (t) of one batch per WG.
//   LDS: A tile 64x1024 bf16 (128KB) + double-buffered B slab 2x32x1024 bf16
//   (2x64KB). B slabs are fetched with GLOBAL_LOAD_ASYNC_TO_LDS_B128
//   (ASYNCcnt) and overlapped with the WMMA compute on the current slab.
// ---------------------------------------------------------------------------
__global__ __launch_bounds__(128)
void k_score(const float* __restrict__ enc,
             const float* __restrict__ dec_w,
             const float* __restrict__ b2,
             const __bf16* __restrict__ W2T,
             const float* __restrict__ V,
             const float* __restrict__ bV,
             float* __restrict__ scores) {
    extern __shared__ char smem[];
    __bf16* Alds  = (__bf16*)smem;                           // [64][1024] 128KB
    __bf16* Bbuf0 = (__bf16*)(smem + 64 * Hh * 2);           // [32][1024]  64KB
    __bf16* Bbuf1 = (__bf16*)(smem + 64 * Hh * 2 + 32 * Hh * 2);

    const int tid  = threadIdx.x;
    const int wave = tid >> 5;
    const int lane = tid & 31;
    const int half = lane >> 4;                         // 0 or 1
    const int lm   = lane & 15;

    const int b  = blockIdx.x >> 5;                     // 32 row-tiles per batch
    const int t0 = (blockIdx.x & 31) * 64;
    const long encBase = ((long)b * Tt + t0) * (long)Hh;

    // Async global->LDS stage of one 32-column B slab (64KB = 4096 x b128).
    // VDST = LDS byte offset (low 32 bits of flat shared ptr), VADDR = 64-bit
    // global address, SADDR = off. Tracked with ASYNCcnt, no VGPR data.
    auto stageB_async = [&](int nbp, __bf16* dst) {
        const char* src = (const char*)(W2T + nbp * 32 * Hh);
        const unsigned dbase = (unsigned)(unsigned long long)(uintptr_t)dst;
        for (int i = tid; i < (32 * Hh * 2) / 16; i += 128) {
            unsigned long long ga = (unsigned long long)(uintptr_t)(src + i * 16);
            unsigned la = dbase + i * 16;
            asm volatile("global_load_async_to_lds_b128 %0, %1, off"
                         :: "v"(la), "v"(ga) : "memory");
        }
    };

    // ---- Stage A: 64 rows x 1024 f32 -> bf16 LDS (once per WG) ----
    {
        const float4* src = (const float4*)(enc + encBase);
        uint2* dst = (uint2*)Alds;
        for (int i = tid; i < (64 * Hh) / 4; i += 128) {
            float4 w = src[i];
            union { __bf16 h[4]; uint2 u; } p;
            p.h[0] = (__bf16)w.x; p.h[1] = (__bf16)w.y;
            p.h[2] = (__bf16)w.z; p.h[3] = (__bf16)w.w;
            dst[i] = p.u;
        }
    }
    // Prefetch first B slab while A conversion drains.
    stageB_async(0, Bbuf0);
    asm volatile("s_wait_asynccnt 0x0" ::: "memory");
    __syncthreads();

    // A operand row for this lane (contiguous k order, not ISA table order: lane -> M=lm,
    // K-chunk {half*16..+15}, same k order as B)
    const __bf16* Arow = Alds + (wave * 16 + lm) * Hh;

    float sacc[8];
#pragma unroll
    for (int v = 0; v < 8; ++v) sacc[v] = 0.f;

    for (int nbp = 0; nbp < LDd / 32; ++nbp) {          // 32 column-pair slabs
        __bf16* Bcur = (nbp & 1) ? Bbuf1 : Bbuf0;
        __bf16* Bnxt = (nbp & 1) ? Bbuf0 : Bbuf1;
        // Kick off async fill of the next slab; overlaps with WMMA below.
        if (nbp + 1 < LDd / 32) stageB_async(nbp + 1, Bnxt);

        // B operand columns for this lane (lane -> N=lm; K-half by lane half)
        const __bf16* Bc0 = Bcur + lm * Hh;
        const __bf16* Bc1 = Bcur + (16 + lm) * Hh;

        v8f acc0 = {}, acc1 = {};
#pragma unroll 4
        for (int ks = 0; ks < Hh / 32; ++ks) {
            const int kb = ks * 32;
            union { v16bf v; uint4 q[2]; } a, b0, b1;
            a.q[0]  = *(const uint4*)(Arow + kb + half * 16);
            a.q[1]  = *(const uint4*)(Arow + kb + half * 16 + 8);
            b0.q[0] = *(const uint4*)(Bc0 + kb + half * 16);
            b0.q[1] = *(const uint4*)(Bc0 + kb + half * 16 + 8);
            b1.q[0] = *(const uint4*)(Bc1 + kb + half * 16);
            b1.q[1] = *(const uint4*)(Bc1 + kb + half * 16 + 8);
            acc0 = __builtin_amdgcn_wmma_f32_16x16x32_bf16(
                false, a.v, false, b0.v, (short)0, acc0, false, false);
            acc1 = __builtin_amdgcn_wmma_f32_16x16x32_bf16(
                false, a.v, false, b1.v, (short)0, acc1, false, false);
        }

        // ---- epilogue: tanh(dec_w * enc_w) * V, accumulated per row ----
        {
            const int n0 = nbp * 32 + lm;               // N of acc0 for this lane
            const int n1 = n0 + 16;                     // N of acc1
            const float dw0  = dec_w[b * LDd + n0];
            const float dw1  = dec_w[b * LDd + n1];
            const float vn0  = V[n0];
            const float vn1  = V[n1];
            const float b2n0 = b2[n0];
            const float b2n1 = b2[n1];
#pragma unroll
            for (int v = 0; v < 8; ++v) {
                sacc[v] += fast_tanh(dw0 * (acc0[v] + b2n0)) * vn0;
                sacc[v] += fast_tanh(dw1 * (acc1[v] + b2n1)) * vn1;
            }
        }
        // Next slab must be fully in LDS (this wave's asyncs done + all waves
        // past their reads of the buffer being refilled next iteration).
        asm volatile("s_wait_asynccnt 0x0" ::: "memory");
        __syncthreads();
    }

    // Reduce over the 16 lanes of each half (they hold N-partials of same rows)
#pragma unroll
    for (int v = 0; v < 8; ++v) {
        float s = sacc[v];
        s += __shfl_xor(s, 1, 32);
        s += __shfl_xor(s, 2, 32);
        s += __shfl_xor(s, 4, 32);
        s += __shfl_xor(s, 8, 32);
        sacc[v] = s;
    }
    if (lm == 0) {
        const float bv = bV[0];
        float* out = scores + b * Tt + t0 + wave * 16 + half * 8;  // rows M=v+8*half
#pragma unroll
        for (int v = 0; v < 8; ++v) out[v] = sacc[v] + bv;
    }
}

// ---------------------------------------------------------------------------
// Kernel 4: softmax over T per batch; writes attn to output
// ---------------------------------------------------------------------------
__global__ void k_softmax(const float* __restrict__ scores,
                          float* __restrict__ attn) {
    __shared__ float red[256];
    const int b = blockIdx.x, tid = threadIdx.x;
    const float* s = scores + b * Tt;
    float vals[Tt / 256];
    float lm = -3.4e38f;
#pragma unroll
    for (int i = 0; i < Tt / 256; ++i) {
        vals[i] = s[tid + i * 256];
        lm = fmaxf(lm, vals[i]);
    }
    red[tid] = lm; __syncthreads();
    for (int off = 128; off > 0; off >>= 1) {
        if (tid < off) red[tid] = fmaxf(red[tid], red[tid + off]);
        __syncthreads();
    }
    const float mx = red[0]; __syncthreads();
    float ls = 0.f;
#pragma unroll
    for (int i = 0; i < Tt / 256; ++i) { vals[i] = __expf(vals[i] - mx); ls += vals[i]; }
    red[tid] = ls; __syncthreads();
    for (int off = 128; off > 0; off >>= 1) {
        if (tid < off) red[tid] += red[tid + off];
        __syncthreads();
    }
    const float inv = 1.f / red[0];
#pragma unroll
    for (int i = 0; i < Tt / 256; ++i) attn[b * Tt + tid + i * 256] = vals[i] * inv;
}

// ---------------------------------------------------------------------------
// Kernel 5: context partials: part[b][c][h] = sum_{t in chunk c} attn*enc
// (atomic-free -> deterministic)
// ---------------------------------------------------------------------------
__global__ void k_ctx_part(const float* __restrict__ enc,
                           const float* __restrict__ attn,
                           float* __restrict__ part) {
    const int blk = blockIdx.x;                 // B*TCH
    const int b = blk / TCH, c = blk % TCH;
    const int tid = threadIdx.x;                // 256
    const int tb = c * (Tt / TCH);
    for (int hb = 0; hb < Hh; hb += 256) {
        const int h = hb + tid;
        float acc = 0.f;
        for (int t = 0; t < Tt / TCH; ++t) {
            const float a = attn[b * Tt + tb + t];             // scalar (uniform)
            acc += a * enc[((long)(b * Tt + tb + t)) * Hh + h]; // coalesced
        }
        part[blk * Hh + h] = acc;
    }
}

// Kernel 6: context[b][h] = sum_c part[b][c][h]
__global__ void k_ctx_reduce(const float* __restrict__ part,
                             float* __restrict__ ctx) {
    const int idx = blockIdx.x * 256 + threadIdx.x;    // B*H
    const int b = idx >> 10, h = idx & (Hh - 1);
    float acc = 0.f;
#pragma unroll
    for (int c = 0; c < TCH; ++c) acc += part[(b * TCH + c) * Hh + h];
    ctx[idx] = acc;
}

// ---------------------------------------------------------------------------
extern "C" void kernel_launch(void* const* d_in, const int* in_sizes, int n_in,
                              void* d_out, int out_size, void* d_ws, size_t ws_size,
                              hipStream_t stream) {
    const float* enc  = (const float*)d_in[0];   // [B,T,H]
    const float* dec  = (const float*)d_in[1];   // [B,H]
    const float* W1   = (const float*)d_in[2];   // [H,LD]
    const float* b1   = (const float*)d_in[3];   // [LD]
    const float* W2   = (const float*)d_in[4];   // [H,LD]
    const float* b2   = (const float*)d_in[5];   // [LD]
    const float* V    = (const float*)d_in[6];   // [LD,1]
    const float* bV   = (const float*)d_in[7];   // [1]

    float* ctx  = (float*)d_out;                         // [B,H]
    float* attn = (float*)d_out + Bsz * Hh;              // [B,T,1]

    // Workspace layout (bytes)
    char* ws = (char*)d_ws;
    float*  dec_w  = (float*)(ws + 0);                   // 128 KB
    float*  scores = (float*)(ws + 131072);              // 256 KB
    float*  part   = (float*)(ws + 393216);              // 1 MB
    __bf16* W2T    = (__bf16*)(ws + 1441792);            // 2 MB

    // 1) decoder projection
    k_decw<<<(Bsz * LDd) / 256, 256, 0, stream>>>(dec, W1, b1, dec_w);
    // 2) W2 transpose+convert to bf16 (B operand layout)
    k_w2t<<<(Hh / 32) * (LDd / 32), 256, 0, stream>>>(W2, W2T);
    // 3) fused GEMM + tanh + V-projection -> scores  (bf16 WMMA, async B)
    const size_t lds_bytes = (size_t)(64 * Hh * 2 + 2 * 32 * Hh * 2);  // 256 KB
    k_score<<<(Bsz * Tt) / 64, 128, lds_bytes, stream>>>(
        enc, dec_w, b2, W2T, V, bV, scores);
    // 4) softmax over T -> attn (output)
    k_softmax<<<Bsz, 256, 0, stream>>>(scores, attn);
    // 5+6) context = sum_t attn*enc (deterministic two-stage)
    k_ctx_part<<<Bsz * TCH, 256, 0, stream>>>(enc, attn, part);
    k_ctx_reduce<<<(Bsz * Hh) / 256, 256, 0, stream>>>(part, ctx);
}
